// OmniAnomalyModel_56556129354318
// MI455X (gfx1250) — matrix operand order repaired and checked by reading, not compiled, not run on hardware
//
#include <hip/hip_runtime.h>
#include <hip/hip_bf16.h>
#include <stdint.h>

// ---------------- types ----------------
typedef __bf16 bf16_t;
typedef bf16_t v16bf __attribute__((ext_vector_type(16)));
typedef float  v8f   __attribute__((ext_vector_type(8)));

union FragAB { v16bf v; uint4 q[2]; };

__device__ __forceinline__ unsigned short f2bf(float f) {
    union { float f; unsigned u; } x; x.f = f;
    unsigned u = x.u;
    unsigned r = (u + 0x7fffu + ((u >> 16) & 1u)) >> 16;   // round-nearest-even
    return (unsigned short)r;
}
__device__ __forceinline__ float bf2f(unsigned short h) {
    union { unsigned u; float f; } x; x.u = ((unsigned)h) << 16;
    return x.f;
}
__device__ __forceinline__ float sigmoidf_(float x) { return 1.0f / (1.0f + __expf(-x)); }
__device__ __forceinline__ float softplusf_(float x) { return (x > 20.0f) ? x : log1pf(__expf(x)); }
__device__ __forceinline__ float leakyf_(float x) { return x > 0.0f ? x : 0.1f * x; }

#define WMMA_BF16(A, B, C) \
    __builtin_amdgcn_wmma_f32_16x16x32_bf16(false, (A), false, (B), (short)0, (C), false, false)

__device__ __forceinline__ void stash8(unsigned short* dst, int colb, int kl, uint4 d) {
    unsigned short hh[8];
    hh[0] = (unsigned short)d.x; hh[1] = (unsigned short)(d.x >> 16);
    hh[2] = (unsigned short)d.y; hh[3] = (unsigned short)(d.y >> 16);
    hh[4] = (unsigned short)d.z; hh[5] = (unsigned short)(d.z >> 16);
    hh[6] = (unsigned short)d.w; hh[7] = (unsigned short)(d.w >> 16);
#pragma unroll
    for (int i = 0; i < 8; ++i) dst[(colb + i) * 40 + kl] = hh[i];
}

// ---------------------------------------------------------------------------
// GRU recurrent step: gates = [h_prev | x_t] @ Wc (+ biases), torch-GRU
// update, writes bf16 h_next and h_seq slice.
// Wc bf16 [512+KX][1536]: cols = 3 gates x 512 (padded). Grid (16, 8).
// Tile: M=32 rows x (3 gates x 64 cols). 8 waves: strip = w>>2, nq = w&3.
// Double-buffered LDS pipeline: one barrier per K-step; next-step global
// loads are issued before the WMMAs of the current step.
// Gate-n accumulator is chosen per K-phase (accNH hidden / accNX input),
// keeping the WMMA region branch-free.
// ---------------------------------------------------------------------------
__global__ __launch_bounds__(256) void gru_step_kernel(
    const unsigned short* __restrict__ hprev,   // [512][512] bf16
    const unsigned short* __restrict__ xin,     // [512][KX]  bf16 (this t)
    const unsigned short* __restrict__ Wc,      // [512+KX][1536] bf16
    const float* __restrict__ bih,              // [1500]
    const float* __restrict__ bhh,              // [1500]
    unsigned short* __restrict__ hnext,         // [512][512]
    unsigned short* __restrict__ hseq,          // [512][512] (slice t)
    int KX)
{
    __shared__ __align__(16) unsigned short As[2][32 * 32];
    __shared__ __align__(16) unsigned short Bt[2][192 * 40];

    const int tid   = threadIdx.x;
    const int lane  = tid & 31;
    const int wave  = tid >> 5;
    const int strip = wave >> 2;     // 0..1 : 16-row strip
    const int nq    = wave & 3;      // 0..3 : 16-col quarter of 64-col tile
    const int m0    = blockIdx.x * 32;
    const int n0    = blockIdx.y * 64;

    // ---- staging geometry (all shift/mask, precomputed once) ----
    const int rA  = tid >> 3;            // A row 0..31
    const int c4  = (tid & 7) << 2;      // A col chunk (4 halves)
    const int klB = tid & 31;            // B k-row 0..31
    const int j0  = tid >> 5;            // B col chunk 0..7
    const unsigned short* Ahp = hprev + (size_t)(m0 + rA) * 512 + c4;
    const unsigned short* Axp = xin   + (size_t)(m0 + rA) * KX  + c4;
    const unsigned short* Wbp = Wc + (size_t)klB * 1536 + n0 + j0 * 8;

    // ---- fragment geometry ----
    const int arow = strip * 16 + (lane & 15);
    const int ka   = (lane < 16) ? 0 : 16;     // A: K 0-15 vs 16-31
    const int kb   = (lane < 16) ? 0 : 16;     // B: K 0-15 vs 16-31
    const int colq = nq * 16 + (lane & 15);

    const v8f vzero = {0.f, 0.f, 0.f, 0.f, 0.f, 0.f, 0.f, 0.f};
    v8f accR = vzero, accZ = vzero, accNH = vzero, accNX = vzero;

    uint2 a;
    uint4 b0, b1, b2;
    auto issue = [&](const unsigned short* Ap, int kbase) {
        a = *(const uint2*)(Ap);
        const unsigned short* w = Wbp + (size_t)kbase * 1536;
        b0 = *(const uint4*)(w);
        b1 = *(const uint4*)(w + 512);
        b2 = *(const uint4*)(w + 1024);
    };
    auto store = [&](int sel) {
        *(uint2*)(As[sel] + rA * 32 + c4) = a;
        stash8(Bt[sel],       j0 * 8, klB, b0);
        stash8(Bt[sel],  64 + j0 * 8, klB, b1);
        stash8(Bt[sel], 128 + j0 * 8, klB, b2);
    };
    auto compute = [&](int sel, v8f& aR, v8f& aZ, v8f& aN) {
        FragAB fa, f0, f1, f2;
        fa.q[0] = *(const uint4*)(As[sel] + arow * 32 + ka);
        fa.q[1] = *(const uint4*)(As[sel] + arow * 32 + ka + 8);
        f0.q[0] = *(const uint4*)(Bt[sel] + (colq)       * 40 + kb);
        f0.q[1] = *(const uint4*)(Bt[sel] + (colq)       * 40 + kb + 8);
        f1.q[0] = *(const uint4*)(Bt[sel] + (64  + colq) * 40 + kb);
        f1.q[1] = *(const uint4*)(Bt[sel] + (64  + colq) * 40 + kb + 8);
        f2.q[0] = *(const uint4*)(Bt[sel] + (128 + colq) * 40 + kb);
        f2.q[1] = *(const uint4*)(Bt[sel] + (128 + colq) * 40 + kb + 8);
        aR = WMMA_BF16(fa.v, f0.v, aR);
        aZ = WMMA_BF16(fa.v, f1.v, aZ);
        aN = WMMA_BF16(fa.v, f2.v, aN);
    };

    const int nx = KX >> 5;   // 2 (encoder) or 1 (decoder)

    // prologue: stage k-step 0
    issue(Ahp, 0);
    store(0);
    __syncthreads();

    // hidden part: K = 0..511 -> accNH (branch-free body, boundary peeled)
    for (int kk = 0; kk < 15; ++kk) {
        issue(Ahp + (kk + 1) * 32, kk + 1);
        compute(kk & 1, accR, accZ, accNH);
        store((kk + 1) & 1);
        __syncthreads();
    }
    // kk = 15: prefetch first x-part tile
    issue(Axp, 16);
    compute(1, accR, accZ, accNH);
    store(0);
    __syncthreads();

    // input part: K = 512.. -> accNX
    for (int kk = 0; kk < nx - 1; ++kk) {
        issue(Axp + (kk + 1) * 32, 17 + kk);
        compute((16 + kk) & 1, accR, accZ, accNX);
        store((17 + kk) & 1);
        __syncthreads();
    }
    compute((16 + nx - 1) & 1, accR, accZ, accNX);   // last step: compute only

    // ---- GRU epilogue ----
    const int colh   = n0 + nq * 16 + (lane & 15);
    const bool valid = colh < 500;
    float br1 = 0.f, bz1 = 0.f, bni = 0.f, br2 = 0.f, bz2 = 0.f, bnh = 0.f;
    if (valid) {
        br1 = bih[colh];         br2 = bhh[colh];
        bz1 = bih[500 + colh];   bz2 = bhh[500 + colh];
        bni = bih[1000 + colh];  bnh = bhh[1000 + colh];
    }
    const int rowBase = m0 + strip * 16 + ((lane >= 16) ? 8 : 0);
#pragma unroll
    for (int v = 0; v < 8; ++v) {
        const int row = rowBase + v;
        float hp = valid ? bf2f(hprev[(size_t)row * 512 + colh]) : 0.0f;
        float r  = sigmoidf_(accR[v] + br1 + br2);
        float zg = sigmoidf_(accZ[v] + bz1 + bz2);
        float nn = tanhf((accNX[v] + bni) + r * (accNH[v] + bnh));
        float hn = (1.0f - zg) * nn + zg * hp;
        unsigned short hb = valid ? f2bf(hn) : (unsigned short)0;
        hnext[(size_t)row * 512 + colh] = hb;
        hseq [(size_t)row * 512 + colh] = hb;
    }
}

// ---------------------------------------------------------------------------
// Post-MLP GEMM: out = leaky_relu(A[M,512] @ W[512,512] + b, 0.1), bf16 out.
// Grid (M/64, 8), 256 threads = 8 waves (4 strips x 2 n-halves).
// ---------------------------------------------------------------------------
__global__ __launch_bounds__(256) void gemm_post_kernel(
    const unsigned short* __restrict__ A,
    const unsigned short* __restrict__ W,
    const float* __restrict__ bias,           // [500]
    unsigned short* __restrict__ out)
{
    __shared__ __align__(16) unsigned short As[64 * 32];
    __shared__ __align__(16) unsigned short Bt[64 * 40];

    const int tid = threadIdx.x, lane = tid & 31, wave = tid >> 5;
    const int strip = wave >> 1, nh = wave & 1;
    const int m0 = blockIdx.x * 64, n0 = blockIdx.y * 64;

    const v8f vzero = {0.f, 0.f, 0.f, 0.f, 0.f, 0.f, 0.f, 0.f};
    v8f acc0 = vzero, acc1 = vzero;

    for (int kk = 0; kk < 16; ++kk) {
        const int k0 = kk << 5;
        {
            const int r = tid >> 2, c8 = (tid & 3) << 3;
            *(uint4*)(As + r * 32 + c8) =
                *(const uint4*)(A + (size_t)(m0 + r) * 512 + k0 + c8);
        }
        {
            const int kl = tid & 31, cho = tid >> 5;   // 256 chunks exactly
            uint4 d = *(const uint4*)(W + (size_t)(k0 + kl) * 512 + n0 + cho * 8);
            stash8(Bt, cho * 8, kl, d);
        }
        __syncthreads();

        FragAB fa;
        const int arow = strip * 16 + (lane & 15);
        const int ka = (lane < 16) ? 0 : 16;
        fa.q[0] = *(const uint4*)(As + arow * 32 + ka);
        fa.q[1] = *(const uint4*)(As + arow * 32 + ka + 8);
        const int kb = (lane < 16) ? 0 : 16;
        const int colb = nh * 32 + (lane & 15);

        FragAB f0, f1;
        f0.q[0] = *(const uint4*)(Bt + (colb)      * 40 + kb);
        f0.q[1] = *(const uint4*)(Bt + (colb)      * 40 + kb + 8);
        f1.q[0] = *(const uint4*)(Bt + (colb + 16) * 40 + kb);
        f1.q[1] = *(const uint4*)(Bt + (colb + 16) * 40 + kb + 8);
        acc0 = WMMA_BF16(fa.v, f0.v, acc0);
        acc1 = WMMA_BF16(fa.v, f1.v, acc1);
        __syncthreads();
    }

    const int rowBase = m0 + strip * 16 + ((lane >= 16) ? 8 : 0);
#pragma unroll
    for (int nt = 0; nt < 2; ++nt) {
        const int col = n0 + nh * 32 + nt * 16 + (lane & 15);
        const bool valid = col < 500;
        const float b = valid ? bias[col] : 0.0f;
#pragma unroll
        for (int v = 0; v < 8; ++v) {
            const float a = nt ? acc1[v] : acc0[v];
            float val = valid ? leakyf_(a + b) : 0.0f;
            out[(size_t)(rowBase + v) * 512 + col] = valid ? f2bf(val) : (unsigned short)0;
        }
    }
}

// ---------------------------------------------------------------------------
// Output heads: mu = A@Wm + bm ; std = softplus(A@Ws + bs) + 1e-4, fp32 out.
// A [51200,512] bf16 (row = t*512 + b). Wm/Ws bf16 [512][48]. Grid (800).
// Waves: strip = w>>1 (rows), which = w&1 (mu vs std), 3 n-tiles cover 48 cols.
// ---------------------------------------------------------------------------
__global__ __launch_bounds__(256) void head_kernel(
    const unsigned short* __restrict__ A,
    const unsigned short* __restrict__ Wm,
    const unsigned short* __restrict__ Ws,
    const float* __restrict__ bm,             // [38]
    const float* __restrict__ bs,             // [38]
    float* __restrict__ outMu,
    float* __restrict__ outStd)
{
    __shared__ __align__(16) unsigned short As[64 * 32];
    __shared__ __align__(16) unsigned short Btm[48 * 40];
    __shared__ __align__(16) unsigned short Bts[48 * 40];

    const int tid = threadIdx.x, lane = tid & 31, wave = tid >> 5;
    const int strip = wave >> 1, which = wave & 1;
    const int m0 = blockIdx.x * 64;

    const v8f vzero = {0.f, 0.f, 0.f, 0.f, 0.f, 0.f, 0.f, 0.f};
    v8f acc0 = vzero, acc1 = vzero, acc2 = vzero;

    for (int kk = 0; kk < 16; ++kk) {
        const int k0 = kk << 5;
        {
            const int r = tid >> 2, c8 = (tid & 3) << 3;
            *(uint4*)(As + r * 32 + c8) =
                *(const uint4*)(A + (size_t)(m0 + r) * 512 + k0 + c8);
        }
        // 384 chunks: c = {tid, tid+256}; mat = c&1, kl = (c>>1)&31, cho = c>>6
        for (int c = tid; c < 384; c += 256) {
            const int mat = c & 1;
            const int rest = c >> 1;
            const int kl  = rest & 31;
            const int cho = rest >> 5;       // 0..5
            const unsigned short* Wsrc = mat ? Ws : Wm;
            uint4 d = *(const uint4*)(Wsrc + (size_t)(k0 + kl) * 48 + cho * 8);
            stash8(mat ? Bts : Btm, cho * 8, kl, d);
        }
        __syncthreads();

        FragAB fa;
        const int arow = strip * 16 + (lane & 15);
        const int ka = (lane < 16) ? 0 : 16;
        fa.q[0] = *(const uint4*)(As + arow * 32 + ka);
        fa.q[1] = *(const uint4*)(As + arow * 32 + ka + 8);
        const int kb = (lane < 16) ? 0 : 16;
        const unsigned short* BT = which ? Bts : Btm;
        const int colb = lane & 15;

        FragAB f0, f1, f2;
        f0.q[0] = *(const uint4*)(BT + (colb)      * 40 + kb);
        f0.q[1] = *(const uint4*)(BT + (colb)      * 40 + kb + 8);
        f1.q[0] = *(const uint4*)(BT + (colb + 16) * 40 + kb);
        f1.q[1] = *(const uint4*)(BT + (colb + 16) * 40 + kb + 8);
        f2.q[0] = *(const uint4*)(BT + (colb + 32) * 40 + kb);
        f2.q[1] = *(const uint4*)(BT + (colb + 32) * 40 + kb + 8);
        acc0 = WMMA_BF16(fa.v, f0.v, acc0);
        acc1 = WMMA_BF16(fa.v, f1.v, acc1);
        acc2 = WMMA_BF16(fa.v, f2.v, acc2);
        __syncthreads();
    }

    const int rowBase = m0 + strip * 16 + ((lane >= 16) ? 8 : 0);
#pragma unroll
    for (int nt = 0; nt < 3; ++nt) {
        const int col = nt * 16 + (lane & 15);
        if (col >= 38) continue;
        const float b = which ? bs[col] : bm[col];
#pragma unroll
        for (int v = 0; v < 8; ++v) {
            const int row = rowBase + v;
            const int bb = row & 511, tt = row >> 9;
            const size_t oi = ((size_t)bb * 100 + tt) * 38 + col;
            const float a = (nt == 0) ? acc0[v] : (nt == 1) ? acc1[v] : acc2[v];
            const float val = a + b;
            if (which) outStd[oi] = softplusf_(val) + 1e-4f;
            else       outMu[oi]  = val;
        }
    }
}

// ---------------------------------------------------------------------------
// Latent step t: mu/std from [h_t | z_prev] (503-dim dot, Z=3). VALU kernel.
// ---------------------------------------------------------------------------
__global__ __launch_bounds__(256) void lat_step_kernel(
    const unsigned short* __restrict__ hpost,  // slice t: [512][512] bf16
    const float* __restrict__ zmW, const float* __restrict__ zmb,
    const float* __restrict__ zsW, const float* __restrict__ zsb,
    const float* __restrict__ eps,
    const float* __restrict__ zprev, float* __restrict__ znext,
    float* __restrict__ outZgen, float* __restrict__ outZmu,
    float* __restrict__ outZstd, int t)
{
    const int idx = blockIdx.x * blockDim.x + threadIdx.x;
    if (idx >= 512 * 3) return;
    const int b = idx / 3, j = idx - b * 3;
    const float* wm = zmW + j * 503;
    const float* ws = zsW + j * 503;
    float sm = 0.f, ss = 0.f;
    for (int i = 0; i < 500; ++i) {
        const float h = bf2f(hpost[(size_t)b * 512 + i]);
        sm += wm[i] * h; ss += ws[i] * h;
    }
#pragma unroll
    for (int i = 0; i < 3; ++i) {
        const float zp = zprev[b * 3 + i];
        sm += wm[500 + i] * zp; ss += ws[500 + i] * zp;
    }
    const float mu = sm + zmb[j];
    const float sd = softplusf_(ss + zsb[j]) + 1e-4f;
    const float e  = eps[((size_t)b * 100 + t) * 3 + j];
    const float z  = mu + sd * e;
    znext[b * 3 + j] = z;
    const size_t oi = ((size_t)b * 100 + t) * 3 + j;
    outZgen[oi] = z; outZmu[oi] = mu; outZstd[oi] = sd;
}

// ---------------------------------------------------------------------------
// Planar flow stack (L=20, Z=3), elementwise over (b,t).
// ---------------------------------------------------------------------------
__global__ __launch_bounds__(256) void flow_kernel(
    const float* __restrict__ zgen,
    const float* __restrict__ fw, const float* __restrict__ fb,
    const float* __restrict__ fu,
    float* __restrict__ outZfin, float* __restrict__ outLd,
    unsigned short* __restrict__ zfbf)         // [T][512][32] bf16
{
    const int idx = blockIdx.x * blockDim.x + threadIdx.x;
    if (idx >= 51200) return;
    const int b = idx / 100, t = idx - b * 100;
    float z0 = zgen[(size_t)idx * 3 + 0];
    float z1 = zgen[(size_t)idx * 3 + 1];
    float z2 = zgen[(size_t)idx * 3 + 2];
    float ld = 0.f;
    for (int l = 0; l < 20; ++l) {
        const float w0 = fw[l * 3], w1 = fw[l * 3 + 1], w2 = fw[l * 3 + 2];
        const float u0 = fu[l * 3], u1 = fu[l * 3 + 1], u2 = fu[l * 3 + 2];
        const float wu = w0 * u0 + w1 * u1 + w2 * u2;
        const float ww = w0 * w0 + w1 * w1 + w2 * w2;
        const float s  = (-1.0f + softplusf_(wu) - wu) / (ww + 1e-7f);
        const float uh0 = u0 + s * w0, uh1 = u1 + s * w1, uh2 = u2 + s * w2;
        const float lin = z0 * w0 + z1 * w1 + z2 * w2 + fb[l];
        const float tt  = tanhf(lin);
        const float det = 1.0f + (1.0f - tt * tt) * (uh0 * w0 + uh1 * w1 + uh2 * w2);
        ld += logf(fabsf(det) + 1e-7f);
        z0 += uh0 * tt; z1 += uh1 * tt; z2 += uh2 * tt;
    }
    outZfin[(size_t)idx * 3 + 0] = z0;
    outZfin[(size_t)idx * 3 + 1] = z1;
    outZfin[(size_t)idx * 3 + 2] = z2;
    outLd[idx] = ld;
    const size_t base = ((size_t)t * 512 + b) * 32;
    zfbf[base + 0] = f2bf(z0); zfbf[base + 1] = f2bf(z1); zfbf[base + 2] = f2bf(z2);
#pragma unroll
    for (int k = 3; k < 32; ++k) zfbf[base + k] = 0;
}

// ---------------------------------------------------------------------------
// Weight prep: out[KP][NG*CO] bf16 = transpose+pad of W[NG*CI][Ksrc] fp32.
// ---------------------------------------------------------------------------
__global__ __launch_bounds__(256) void transpose_pad_kernel(
    const float* __restrict__ W, unsigned short* __restrict__ out,
    int KP, int NG, int CI, int CO, int Ksrc)
{
    const int total = KP * NG * CO;
    const int idx = blockIdx.x * blockDim.x + threadIdx.x;
    if (idx >= total) return;
    const int n = idx % (NG * CO);
    const int k = idx / (NG * CO);
    const int g = n / CO, c = n - g * CO;
    float v = 0.0f;
    if (c < CI && k < Ksrc) v = W[((size_t)g * CI + c) * Ksrc + k];
    out[(size_t)k * (NG * CO) + n] = f2bf(v);
}

__global__ __launch_bounds__(256) void pack_x_kernel(
    const float* __restrict__ x, unsigned short* __restrict__ xb)
{
    const int idx = blockIdx.x * blockDim.x + threadIdx.x;
    if (idx >= 100 * 512 * 64) return;
    const int c = idx & 63;
    const int rb = idx >> 6;            // rb = t*512 + b
    const int b = rb & 511, t = rb >> 9;
    const float v = (c < 38) ? x[((size_t)b * 100 + t) * 38 + c] : 0.0f;
    xb[idx] = f2bf(v);
}

__global__ __launch_bounds__(256) void zero_u16_kernel(unsigned short* p, int n) {
    const int i = blockIdx.x * blockDim.x + threadIdx.x;
    if (i < n) p[i] = 0;
}
__global__ __launch_bounds__(256) void zero_f32_kernel(float* p, int n) {
    const int i = blockIdx.x * blockDim.x + threadIdx.x;
    if (i < n) p[i] = 0.0f;
}

// ---------------------------------------------------------------------------
extern "C" void kernel_launch(void* const* d_in, const int* in_sizes, int n_in,
                              void* d_out, int out_size, void* d_ws, size_t ws_size,
                              hipStream_t stream)
{
    (void)in_sizes; (void)n_in; (void)out_size; (void)ws_size;
    const float* x       = (const float*)d_in[0];
    const float* eps     = (const float*)d_in[1];
    const float* enc_Wih = (const float*)d_in[2];
    const float* enc_Whh = (const float*)d_in[3];
    const float* enc_bih = (const float*)d_in[4];
    const float* enc_bhh = (const float*)d_in[5];
    const float* enc_W1  = (const float*)d_in[6];
    const float* enc_b1  = (const float*)d_in[7];
    const float* enc_W2  = (const float*)d_in[8];
    const float* enc_b2  = (const float*)d_in[9];
    const float* zm_W    = (const float*)d_in[10];
    const float* zm_b    = (const float*)d_in[11];
    const float* zs_W    = (const float*)d_in[12];
    const float* zs_b    = (const float*)d_in[13];
    const float* flow_w  = (const float*)d_in[14];
    const float* flow_b  = (const float*)d_in[15];
    const float* flow_u  = (const float*)d_in[16];
    const float* dec_Wih = (const float*)d_in[17];
    const float* dec_Whh = (const float*)d_in[18];
    const float* dec_bih = (const float*)d_in[19];
    const float* dec_bhh = (const float*)d_in[20];
    const float* dec_W1  = (const float*)d_in[21];
    const float* dec_b1  = (const float*)d_in[22];
    const float* dec_W2  = (const float*)d_in[23];
    const float* dec_b2  = (const float*)d_in[24];
    const float* xm_W    = (const float*)d_in[25];
    const float* xm_b    = (const float*)d_in[26];
    const float* xs_W    = (const float*)d_in[27];
    const float* xs_b    = (const float*)d_in[28];

    char* ws = (char*)d_ws;
    size_t off = 0;
    auto alloc = [&](size_t bytes) -> char* {
        char* p = ws + off;
        off = (off + bytes + 255) & ~(size_t)255;
        return p;
    };
    unsigned short* WcE  = (unsigned short*)alloc((size_t)576 * 1536 * 2);
    unsigned short* WcD  = (unsigned short*)alloc((size_t)544 * 1536 * 2);
    unsigned short* W1e  = (unsigned short*)alloc((size_t)512 * 512 * 2);
    unsigned short* W2e  = (unsigned short*)alloc((size_t)512 * 512 * 2);
    unsigned short* W1d  = (unsigned short*)alloc((size_t)512 * 512 * 2);
    unsigned short* W2d  = (unsigned short*)alloc((size_t)512 * 512 * 2);
    unsigned short* Wxm  = (unsigned short*)alloc((size_t)512 * 48 * 2);
    unsigned short* Wxs  = (unsigned short*)alloc((size_t)512 * 48 * 2);
    unsigned short* xbf  = (unsigned short*)alloc((size_t)100 * 512 * 64 * 2);
    unsigned short* hA   = (unsigned short*)alloc((size_t)512 * 512 * 2);
    unsigned short* hB   = (unsigned short*)alloc((size_t)512 * 512 * 2);
    unsigned short* seq  = (unsigned short*)alloc((size_t)51200 * 512 * 2);
    unsigned short* mid  = (unsigned short*)alloc((size_t)51200 * 512 * 2);
    unsigned short* post = (unsigned short*)alloc((size_t)51200 * 512 * 2);
    unsigned short* zfbf = (unsigned short*)alloc((size_t)100 * 512 * 32 * 2);
    float* zcA = (float*)alloc((size_t)512 * 3 * 4);
    float* zcB = (float*)alloc((size_t)512 * 3 * 4);

    float* out     = (float*)d_out;
    float* outMu   = out;
    float* outStd  = out + 1945600;
    float* outZgen = out + 3891200;
    float* outZfin = out + 4044800;
    float* outZmu  = out + 4198400;
    float* outZstd = out + 4352000;
    float* outLd   = out + 4505600;

    auto g1 = [](int n) { return dim3((unsigned)((n + 255) / 256)); };

    // ---- weight prep ----
    transpose_pad_kernel<<<g1(512 * 1536), 256, 0, stream>>>(enc_Whh, WcE, 512, 3, 500, 512, 500);
    transpose_pad_kernel<<<g1(64 * 1536), 256, 0, stream>>>(enc_Wih, WcE + (size_t)512 * 1536, 64, 3, 500, 512, 38);
    transpose_pad_kernel<<<g1(512 * 1536), 256, 0, stream>>>(dec_Whh, WcD, 512, 3, 500, 512, 500);
    transpose_pad_kernel<<<g1(32 * 1536), 256, 0, stream>>>(dec_Wih, WcD + (size_t)512 * 1536, 32, 3, 500, 512, 3);
    transpose_pad_kernel<<<g1(512 * 512), 256, 0, stream>>>(enc_W1, W1e, 512, 1, 500, 512, 500);
    transpose_pad_kernel<<<g1(512 * 512), 256, 0, stream>>>(enc_W2, W2e, 512, 1, 500, 512, 500);
    transpose_pad_kernel<<<g1(512 * 512), 256, 0, stream>>>(dec_W1, W1d, 512, 1, 500, 512, 500);
    transpose_pad_kernel<<<g1(512 * 512), 256, 0, stream>>>(dec_W2, W2d, 512, 1, 500, 512, 500);
    transpose_pad_kernel<<<g1(512 * 48), 256, 0, stream>>>(xm_W, Wxm, 512, 1, 38, 48, 500);
    transpose_pad_kernel<<<g1(512 * 48), 256, 0, stream>>>(xs_W, Wxs, 512, 1, 38, 48, 500);
    pack_x_kernel<<<g1(100 * 512 * 64), 256, 0, stream>>>(x, xbf);

    // ---- encoder GRU ----
    zero_u16_kernel<<<g1(512 * 512), 256, 0, stream>>>(hA, 512 * 512);
    unsigned short* cur = hA; unsigned short* nxt = hB;
    for (int t = 0; t < 100; ++t) {
        gru_step_kernel<<<dim3(16, 8), 256, 0, stream>>>(
            cur, xbf + (size_t)t * 512 * 64, WcE, enc_bih, enc_bhh,
            nxt, seq + (size_t)t * 512 * 512, 64);
        unsigned short* tmp = cur; cur = nxt; nxt = tmp;
    }
    gemm_post_kernel<<<dim3(800, 8), 256, 0, stream>>>(seq, W1e, enc_b1, mid);
    gemm_post_kernel<<<dim3(800, 8), 256, 0, stream>>>(mid, W2e, enc_b2, post);

    // ---- latent scan ----
    zero_f32_kernel<<<g1(512 * 3), 256, 0, stream>>>(zcA, 512 * 3);
    float* zc = zcA; float* zn = zcB;
    for (int t = 0; t < 100; ++t) {
        lat_step_kernel<<<g1(512 * 3), 256, 0, stream>>>(
            post + (size_t)t * 512 * 512, zm_W, zm_b, zs_W, zs_b, eps,
            zc, zn, outZgen, outZmu, outZstd, t);
        float* tmp = zc; zc = zn; zn = tmp;
    }

    // ---- planar flows ----
    flow_kernel<<<g1(51200), 256, 0, stream>>>(outZgen, flow_w, flow_b, flow_u,
                                               outZfin, outLd, zfbf);

    // ---- decoder GRU ----
    zero_u16_kernel<<<g1(512 * 512), 256, 0, stream>>>(hA, 512 * 512);
    cur = hA; nxt = hB;
    for (int t = 0; t < 100; ++t) {
        gru_step_kernel<<<dim3(16, 8), 256, 0, stream>>>(
            cur, zfbf + (size_t)t * 512 * 32, WcD, dec_bih, dec_bhh,
            nxt, seq + (size_t)t * 512 * 512, 32);
        unsigned short* tmp = cur; cur = nxt; nxt = tmp;
    }
    gemm_post_kernel<<<dim3(800, 8), 256, 0, stream>>>(seq, W1d, dec_b1, mid);
    gemm_post_kernel<<<dim3(800, 8), 256, 0, stream>>>(mid, W2d, dec_b2, post);

    // ---- output heads ----
    head_kernel<<<dim3(800), 256, 0, stream>>>(post, Wxm, Wxs, xm_b, xs_b, outMu, outStd);
}
